// Mamba_83769042141785
// MI455X (gfx1250) — compile-verified, matrix operand order checked by reading, not run on hardware
//
#include <hip/hip_runtime.h>
#include <hip/hip_bf16.h>
#include <math.h>

// ---------------------------------------------------------------------------
// Types for CDNA5 WMMA (wave32)
// ---------------------------------------------------------------------------
typedef __bf16 bf16_t;
typedef bf16_t v4bf  __attribute__((ext_vector_type(4)));
typedef bf16_t v8bf  __attribute__((ext_vector_type(8)));
typedef bf16_t v16bf __attribute__((ext_vector_type(16)));
typedef float  v8f   __attribute__((ext_vector_type(8)));

// Problem constants
#define D_MODEL  1024
#define D_INNER  2048
#define D_STATE  16
#define D_CONV   4
#define DT_RANK  64
#define BATCH    4
#define SEQL     2048
#define ROWS     (BATCH * SEQL)      // 8192  (M for every GEMM)

// Chunked scan
#define CHUNK    128
#define NCHUNK   (SEQL / CHUNK)      // 16

// GEMM tiling
#define BM  128
#define BN  64
#define BK  32
#define BKP 40                       // padded LDS row (80 B): 16B-aligned chunks, conflict-free

__device__ __forceinline__ float siluf(float x)     { return x / (1.0f + __expf(-x)); }
__device__ __forceinline__ float softplusf(float x) { return (x > 20.0f) ? x : log1pf(__expf(x)); }

__device__ __forceinline__ v4bf cvt4(float4 v) {
    v4bf r; r[0] = (bf16_t)v.x; r[1] = (bf16_t)v.y; r[2] = (bf16_t)v.z; r[3] = (bf16_t)v.w;
    return r;
}

// ---------------------------------------------------------------------------
// Staging helpers: vectorized float4 global loads into registers.
// A tile: BM x BK = 4096 floats -> 4 float4 per thread (256 threads)
// W tile: BN x BK = 2048 floats -> 2 float4 per thread
// M and K are always multiples of the tile; only N may be ragged (NB=true).
// ---------------------------------------------------------------------------
__device__ __forceinline__ void load_A_tile(const float* __restrict__ A, int lda,
                                            int rowBase, int kt, int tid, float4 ra[4]) {
#pragma unroll
    for (int it = 0; it < 4; ++it) {
        int q = tid + 256 * it;
        int r = q >> 3, kk = (q & 7) * 4;
        ra[it] = *(const float4*)(A + (size_t)(rowBase + r) * lda + kt + kk);
    }
}

template <bool NB>
__device__ __forceinline__ void load_W_tile(const float* __restrict__ W, int ldw,
                                            int colBase, int N, int kt, int tid, float4 rb[2]) {
#pragma unroll
    for (int it = 0; it < 2; ++it) {
        int q = tid + 256 * it;
        int n = q >> 3, kk = (q & 7) * 4;
        if (!NB || (colBase + n) < N)
            rb[it] = *(const float4*)(W + (size_t)(colBase + n) * ldw + kt + kk);
        else
            rb[it] = make_float4(0.f, 0.f, 0.f, 0.f);
    }
}

// ---------------------------------------------------------------------------
// Tiled GEMM:  C[M,N] = A[M,K] * W[N,K]^T   (fp32 global, bf16 WMMA, f32 acc)
// mode 0: plain store;  mode 1: softplus(v + bias[n])  (dt projection)
// 256 threads = 8 waves; each wave computes a 16(M) x 64(N) strip:
//   1 A-fragment reused across 4 B-fragments -> 4 v_wmma per K-step.
// Register double-buffered staging overlaps global loads with WMMA.
// ---------------------------------------------------------------------------
template <bool NB>
__global__ __launch_bounds__(256)
void gemm_wmma_bf16(const float* __restrict__ A, int lda,
                    const float* __restrict__ W, int ldw,
                    float* __restrict__ C, int ldc,
                    int N, int K,
                    int mode, const float* __restrict__ bias) {
    __shared__ __align__(16) bf16_t As[BM][BKP];
    __shared__ __align__(16) bf16_t Bs[BN][BKP];

    const int tid     = threadIdx.x;
    const int wave    = tid >> 5;
    const int lane    = tid & 31;
    const int half    = lane >> 4;      // 0: lanes 0-15, 1: lanes 16-31
    const int m       = lane & 15;
    const int kb      = half * 8;       // per-lane K base within BK=32 fragment
    const int rowBase = blockIdx.y * BM;
    const int colBase = blockIdx.x * BN;

    v8f acc[4];
#pragma unroll
    for (int j = 0; j < 4; ++j)
#pragma unroll
        for (int r = 0; r < 8; ++r) acc[j][r] = 0.0f;

    float4 ra[4], rb[2];
    load_A_tile(A, lda, rowBase, 0, tid, ra);
    load_W_tile<NB>(W, ldw, colBase, N, 0, tid, rb);

    for (int kt = 0; kt < K; kt += BK) {
        // ---- write staged registers to LDS (bf16) ----
#pragma unroll
        for (int it = 0; it < 4; ++it) {
            int q = tid + 256 * it;
            int r = q >> 3, kk = (q & 7) * 4;
            *(v4bf*)&As[r][kk] = cvt4(ra[it]);
        }
#pragma unroll
        for (int it = 0; it < 2; ++it) {
            int q = tid + 256 * it;
            int n = q >> 3, kk = (q & 7) * 4;
            *(v4bf*)&Bs[n][kk] = cvt4(rb[it]);
        }
        __syncthreads();

        // ---- kick off next tile's global loads (overlap with WMMA) ----
        const bool more = (kt + BK) < K;
        if (more) {
            load_A_tile(A, lda, rowBase, kt + BK, tid, ra);
            load_W_tile<NB>(W, ldw, colBase, N, kt + BK, tid, rb);
        }

        // ---- A fragment: 16x32, lane m holds row (wave*16+m) ----
        v8bf alo = *(const v8bf*)&As[wave * 16 + m][kb];
        v8bf ahi = *(const v8bf*)&As[wave * 16 + m][kb + 16];
        v16bf afrag = __builtin_shufflevector(alo, ahi,
            0,1,2,3,4,5,6,7,8,9,10,11,12,13,14,15);

#pragma unroll
        for (int j = 0; j < 4; ++j) {
            v8bf blo = *(const v8bf*)&Bs[j * 16 + m][kb];
            v8bf bhi = *(const v8bf*)&Bs[j * 16 + m][kb + 16];
            v16bf bfrag = __builtin_shufflevector(blo, bhi,
                0,1,2,3,4,5,6,7,8,9,10,11,12,13,14,15);
            acc[j] = __builtin_amdgcn_wmma_f32_16x16x32_bf16(
                false, afrag, false, bfrag, (short)0, acc[j], false, false);
        }
        __syncthreads();
    }

    // ---- epilogue + store (C/D layout: VGPR r -> M = 16*wave + r + 8*half) ----
#pragma unroll
    for (int j = 0; j < 4; ++j) {
        int gn = colBase + j * 16 + m;
        if (NB && gn >= N) continue;
#pragma unroll
        for (int r = 0; r < 8; ++r) {
            int gm = rowBase + wave * 16 + half * 8 + r;
            float v = acc[j][r];
            if (mode == 1) v = softplusf(v + bias[gn]);
            C[(size_t)gm * ldc + gn] = v;
        }
    }
}

// ---------------------------------------------------------------------------
// Causal depthwise conv1d (width 4) + bias + SiLU.
// x, xc layout: (b*L + l, d_inner) row-major -> coalesced in d.
// ---------------------------------------------------------------------------
__global__ __launch_bounds__(256)
void conv_silu_kernel(const float* __restrict__ x,
                      const float* __restrict__ cw,
                      const float* __restrict__ cb,
                      float* __restrict__ xc) {
    int idx = blockIdx.x * 256 + threadIdx.x;
    int d   = idx & (D_INNER - 1);
    int row = idx >> 11;                 // D_INNER == 2048
    int l   = row & (SEQL - 1);          // L == 2048

    float acc = cb[d];
#pragma unroll
    for (int j = 0; j < D_CONV; ++j) {
        int lj = l - (D_CONV - 1) + j;
        if (lj >= 0)
            acc += x[(size_t)(row - (D_CONV - 1) + j) * D_INNER + d] * cw[d * D_CONV + j];
    }
    xc[(size_t)row * D_INNER + d] = siluf(acc);
}

// ---------------------------------------------------------------------------
// Chunked selective scan (linear recurrence h' = dA*h + dBx*B):
//  pass1: per chunk, h0=0 -> store chunk end-state and decay product
//  pass2: per (b,d), 16-step sequential fixup -> chunk start states (in place)
//  pass3: per chunk, replay from corrected start state, emit y
// Thread mapping keeps d consecutive within a wave -> coalesced x/delta/z.
// xdbl layout: (row, 96) = [dt(64) | B(16) | C(16)]  (wave-uniform broadcasts)
// ---------------------------------------------------------------------------
__global__ __launch_bounds__(256)
void scan_pass1(const float* __restrict__ xc,
                const float* __restrict__ xdbl,
                const float* __restrict__ A_log,
                const float* __restrict__ dy,       // delta
                float* __restrict__ hend,           // [b][c][d][16]
                float* __restrict__ prod) {         // [b][c][d][16]
    int t = blockIdx.x * 256 + threadIdx.x;         // BATCH*NCHUNK*D_INNER threads
    int d = t & (D_INNER - 1);
    int c = (t >> 11) & (NCHUNK - 1);
    int b = t >> 15;

    float a[D_STATE], h[D_STATE], p[D_STATE];
#pragma unroll
    for (int n = 0; n < D_STATE; ++n) {
        a[n] = -__expf(A_log[d * D_STATE + n]);
        h[n] = 0.0f;
        p[n] = 1.0f;
    }

    size_t base   = (size_t)b * SEQL * D_INNER + (size_t)c * CHUNK * D_INNER + d;
    size_t bcrow  = (size_t)b * SEQL * 96 + (size_t)c * CHUNK * 96;

    for (int l = 0; l < CHUNK; ++l) {
        size_t off = base + (size_t)l * D_INNER;
        float xv = xc[off];
        float dl = dy[off];
        const float* Bp = xdbl + bcrow + (size_t)l * 96 + DT_RANK;
        float dBx = dl * xv;
#pragma unroll
        for (int n = 0; n < D_STATE; ++n) {
            float dA = __expf(dl * a[n]);
            h[n] = dA * h[n] + dBx * Bp[n];
            p[n] *= dA;
        }
    }

    size_t s = ((size_t)(b * NCHUNK + c) * D_INNER + d) * D_STATE;
#pragma unroll
    for (int i = 0; i < 4; ++i) {
        *(float4*)&hend[s + 4 * i] = make_float4(h[4*i], h[4*i+1], h[4*i+2], h[4*i+3]);
        *(float4*)&prod[s + 4 * i] = make_float4(p[4*i], p[4*i+1], p[4*i+2], p[4*i+3]);
    }
}

__global__ __launch_bounds__(256)
void scan_pass2(float* __restrict__ hend,           // in: end states; out: START states
                const float* __restrict__ prod) {
    int t = blockIdx.x * 256 + threadIdx.x;         // BATCH*D_INNER threads
    int d = t & (D_INNER - 1);
    int b = t >> 11;

    float H[D_STATE];
#pragma unroll
    for (int n = 0; n < D_STATE; ++n) H[n] = 0.0f;

    for (int c = 0; c < NCHUNK; ++c) {
        size_t s = ((size_t)(b * NCHUNK + c) * D_INNER + d) * D_STATE;
#pragma unroll
        for (int n = 0; n < D_STATE; ++n) {
            float e = hend[s + n];
            float p = prod[s + n];
            hend[s + n] = H[n];                     // start state for chunk c
            H[n] = p * H[n] + e;
        }
    }
}

__global__ __launch_bounds__(256)
void scan_pass3(const float* __restrict__ xc,
                const float* __restrict__ xdbl,
                const float* __restrict__ zbuf,
                const float* __restrict__ A_log,
                const float* __restrict__ Dskip,
                const float* __restrict__ hstart,   // [b][c][d][16]
                float* __restrict__ dy) {           // delta in, y out (in place)
    int t = blockIdx.x * 256 + threadIdx.x;
    int d = t & (D_INNER - 1);
    int c = (t >> 11) & (NCHUNK - 1);
    int b = t >> 15;

    float a[D_STATE], h[D_STATE];
    size_t s = ((size_t)(b * NCHUNK + c) * D_INNER + d) * D_STATE;
#pragma unroll
    for (int n = 0; n < D_STATE; ++n) {
        a[n] = -__expf(A_log[d * D_STATE + n]);
        h[n] = hstart[s + n];
    }
    float Dd = Dskip[d];

    size_t base  = (size_t)b * SEQL * D_INNER + (size_t)c * CHUNK * D_INNER + d;
    size_t bcrow = (size_t)b * SEQL * 96 + (size_t)c * CHUNK * 96;

    for (int l = 0; l < CHUNK; ++l) {
        size_t off = base + (size_t)l * D_INNER;
        float xv = xc[off];
        float dl = dy[off];
        float zv = zbuf[off];
        const float* Bp = xdbl + bcrow + (size_t)l * 96 + DT_RANK;
        const float* Cp = Bp + D_STATE;
        float dBx = dl * xv;
        float y = 0.0f;
#pragma unroll
        for (int n = 0; n < D_STATE; ++n) {
            float dA = __expf(dl * a[n]);
            h[n] = dA * h[n] + dBx * Bp[n];
            y += h[n] * Cp[n];
        }
        y += Dd * xv;
        dy[off] = y * siluf(zv);
    }
}

// ---------------------------------------------------------------------------
// Host-side launcher.
// ---------------------------------------------------------------------------
static void run_branch(const float* hs,
                       const float* in_w, const float* conv_w, const float* conv_b,
                       const float* xproj_w, const float* dt_w, const float* dt_b,
                       const float* A_log, const float* Dskip, const float* out_w,
                       float* out, float* ws, hipStream_t stream) {
    // workspace layout (floats)
    float* xbuf = ws;                               // ROWS * D_INNER
    float* zbuf = xbuf + (size_t)ROWS * D_INNER;    // ROWS * D_INNER
    float* xc   = zbuf + (size_t)ROWS * D_INNER;    // ROWS * D_INNER
    float* xdbl = xc   + (size_t)ROWS * D_INNER;    // ROWS * 96
    float* dy   = xdbl + (size_t)ROWS * 96;         // ROWS * D_INNER (delta then y)
    float* hend = dy   + (size_t)ROWS * D_INNER;    // BATCH*NCHUNK*D_INNER*16
    float* prod = hend + (size_t)BATCH * NCHUNK * D_INNER * D_STATE;

    dim3 blk(256);

    // 1) in_proj -> x half (W rows [0,2048))
    gemm_wmma_bf16<false><<<dim3(D_INNER / BN, ROWS / BM), blk, 0, stream>>>(
        hs, D_MODEL, in_w, D_MODEL, xbuf, D_INNER, D_INNER, D_MODEL, 0, nullptr);
    // 2) in_proj -> z half (W rows [2048,4096))
    gemm_wmma_bf16<false><<<dim3(D_INNER / BN, ROWS / BM), blk, 0, stream>>>(
        hs, D_MODEL, in_w + (size_t)D_INNER * D_MODEL, D_MODEL, zbuf, D_INNER,
        D_INNER, D_MODEL, 0, nullptr);
    // 3) causal depthwise conv + SiLU
    conv_silu_kernel<<<(ROWS * D_INNER) / 256, blk, 0, stream>>>(xbuf, conv_w, conv_b, xc);
    // 4) x_proj: ragged N=96 -> bounded variant
    gemm_wmma_bf16<true><<<dim3((96 + BN - 1) / BN, ROWS / BM), blk, 0, stream>>>(
        xc, D_INNER, xproj_w, D_INNER, xdbl, 96, 96, D_INNER, 0, nullptr);
    // 5) dt_proj: delta = softplus(xdbl[:, :64] * dt_w^T + dt_b)
    gemm_wmma_bf16<false><<<dim3(D_INNER / BN, ROWS / BM), blk, 0, stream>>>(
        xdbl, 96, dt_w, DT_RANK, dy, D_INNER, D_INNER, DT_RANK, 1, dt_b);
    // 6) chunked selective scan
    scan_pass1<<<(BATCH * NCHUNK * D_INNER) / 256, blk, 0, stream>>>(
        xc, xdbl, A_log, dy, hend, prod);
    scan_pass2<<<(BATCH * D_INNER) / 256, blk, 0, stream>>>(hend, prod);
    scan_pass3<<<(BATCH * NCHUNK * D_INNER) / 256, blk, 0, stream>>>(
        xc, xdbl, zbuf, A_log, Dskip, hend, dy);
    // 7) out_proj
    gemm_wmma_bf16<false><<<dim3(D_MODEL / BN, ROWS / BM), blk, 0, stream>>>(
        dy, D_INNER, out_w, D_INNER, out, D_MODEL, D_MODEL, D_INNER, 0, nullptr);
}

extern "C" void kernel_launch(void* const* d_in, const int* in_sizes, int n_in,
                              void* d_out, int out_size, void* d_ws, size_t ws_size,
                              hipStream_t stream) {
    (void)in_sizes; (void)n_in; (void)out_size; (void)ws_size;

    const float* hs_a = (const float*)d_in[0];
    const float* hs_v = (const float*)d_in[1];
    float* out = (float*)d_out;
    float* ws  = (float*)d_ws;

    run_branch(hs_a,
               (const float*)d_in[2], (const float*)d_in[3], (const float*)d_in[4],
               (const float*)d_in[5], (const float*)d_in[6], (const float*)d_in[7],
               (const float*)d_in[8], (const float*)d_in[9], (const float*)d_in[10],
               out, ws, stream);
    run_branch(hs_v,
               (const float*)d_in[11], (const float*)d_in[12], (const float*)d_in[13],
               (const float*)d_in[14], (const float*)d_in[15], (const float*)d_in[16],
               (const float*)d_in[17], (const float*)d_in[18], (const float*)d_in[19],
               out + (size_t)ROWS * D_MODEL, ws, stream);
}
